// VGATEncoder_48876727828951
// MI455X (gfx1250) — matrix operand order repaired and checked by reading, not compiled, not run on hardware
//
#include <hip/hip_runtime.h>
#include <math.h>

#define DIN 128
#define DH  128
#define DZ  64
#define SLOPE 0.2f

typedef __attribute__((ext_vector_type(16))) _Float16 v16h;
typedef __attribute__((ext_vector_type(8)))  float    v8f;

#if defined(__gfx1250__)
#define USE_ASYNC_LDS 1
#endif

// ---------------- helpers ----------------

__device__ __forceinline__ float waveReduceSum(float v) {
#pragma unroll
    for (int off = 16; off > 0; off >>= 1) v += __shfl_down(v, off, 32);
    return v;
}

// float atomic max via monotonic int/uint ordering (dest initialized to -inf)
__device__ __forceinline__ void atomicMaxF(float* addr, float v) {
    if (v >= 0.0f) atomicMax((int*)addr, __float_as_int(v));
    else           atomicMin((unsigned int*)addr, __float_as_uint(v));
}

__device__ __forceinline__ void edge_nodes(const int* __restrict__ ei, int E,
                                           int e, int& src, int& dst) {
    if (e < E) { src = ei[e]; dst = ei[E + e]; }   // edge_index rows: [0]=src, [1]=dst
    else       { src = dst = e - E; }              // self loops appended
}

// ---------------- generic fill ----------------

__global__ void fill_kernel(float* __restrict__ p, float v, long long n) {
    long long i = (long long)blockIdx.x * blockDim.x + threadIdx.x;
    if (i < n) p[i] = v;
}

// ---------------- WMMA GEMM: C[MxNcol] = A[MxK] * B[KxNcol] ----------------
// fp32 in, f16 WMMA w/ f32 accumulate, fp32 out. One wave computes a 16x32 C
// slab (two 16x16 tiles sharing one A fragment -> 2 WMMAs per A load).
// B is async-copied (GLOBAL_LOAD_ASYNC_TO_LDS_B128 / ASYNCcnt, inline asm: the
// clang builtin wants unspellable __device__/__shared__ AS pointer params),
// then converted once to f16 column-major so fragments are contiguous
// ds_load_b128 in the inner loop.

template <int K>
__global__ __launch_bounds__(256)
void gemm_wmma_kernel(const float* __restrict__ A, const float* __restrict__ B,
                      float* __restrict__ C, int M, int Ncol) {
    __shared__ __align__(16) float    sBf[K * 128];   // raw fp32 stage (<=64 KB)
    __shared__ __align__(16) _Float16 sBh[K * 128];   // f16 col-major (<=32 KB)
    const int tid   = threadIdx.x;
    const int total = K * Ncol;                        // multiple of 1024

#if defined(USE_ASYNC_LDS)
    for (int c = tid * 4; c < total; c += 256 * 4) {
        const float* gsrc = B + c;
        unsigned ldsa = (unsigned)(size_t)(sBf + c);   // LDS byte addr = low 32 bits
        asm volatile("global_load_async_to_lds_b128 %0, %1, off"
                     :: "v"(ldsa), "v"(gsrc) : "memory");
    }
    asm volatile("s_wait_asynccnt 0" ::: "memory");
#else
    for (int idx = tid; idx < total; idx += 256) sBf[idx] = B[idx];
#endif
    __syncthreads();
    for (int idx = tid; idx < total; idx += 256) {
        int k = idx / Ncol, nn = idx % Ncol;
        sBh[nn * K + k] = (_Float16)sBf[idx];          // transpose + convert once
    }
    __syncthreads();

    const int lane    = tid & 31;
    const int wid     = (blockIdx.x * 256 + tid) >> 5;
    const int tilesN2 = Ncol >> 5;                     // N-tile *pairs*
    const int tilesM  = M >> 4;
    const int tm = wid / tilesN2;
    const int tq = wid % tilesN2;
    if (tm >= tilesM) return;

    // A frag (16x32 f16): lane L holds row M=L%16; K = (L<16?0:16) + i
    // B frag (32x16 f16): lane L holds col N=L%16; K = (L<16?0:16) + i
    const int row  = (tm << 4) + (lane & 15);
    const int aOff = (lane < 16) ? 0 : 16;
    const int bOff = (lane < 16) ? 0 : 16;
    const int bN   = (tq << 5) + (lane & 15);
    const float*    aRow  = A + (size_t)row * K;
    const _Float16* bCol0 = &sBh[bN * K];
    const _Float16* bCol1 = &sBh[(bN + 16) * K];

    v8f acc0 = {}, acc1 = {};
#pragma unroll
    for (int k0 = 0; k0 < K; k0 += 32) {
        const float* ap = aRow + k0 + aOff;
        float4 a0 = *(const float4*)(ap);
        float4 a1 = *(const float4*)(ap + 4);
        float4 a2 = *(const float4*)(ap + 8);
        float4 a3 = *(const float4*)(ap + 12);
        v16h a;
        a[0]=(_Float16)a0.x;  a[1]=(_Float16)a0.y;  a[2]=(_Float16)a0.z;  a[3]=(_Float16)a0.w;
        a[4]=(_Float16)a1.x;  a[5]=(_Float16)a1.y;  a[6]=(_Float16)a1.z;  a[7]=(_Float16)a1.w;
        a[8]=(_Float16)a2.x;  a[9]=(_Float16)a2.y;  a[10]=(_Float16)a2.z; a[11]=(_Float16)a2.w;
        a[12]=(_Float16)a3.x; a[13]=(_Float16)a3.y; a[14]=(_Float16)a3.z; a[15]=(_Float16)a3.w;

        const _Float16* bp0 = bCol0 + k0 + bOff;
        const _Float16* bp1 = bCol1 + k0 + bOff;
        v16h b0, b1;
#pragma unroll
        for (int i = 0; i < 16; ++i) { b0[i] = bp0[i]; b1[i] = bp1[i]; }

        acc0 = __builtin_amdgcn_wmma_f32_16x16x32_f16(false, a, false, b0,
                                                      (short)0, acc0, false, false);
        acc1 = __builtin_amdgcn_wmma_f32_16x16x32_f16(false, a, false, b1,
                                                      (short)0, acc1, false, false);
    }

    // C/D layout: VGPR r -> M = r + (lane<16?0:8), N = lane%16
    const int cRow = (tm << 4) + ((lane < 16) ? 0 : 8);
    const int cCol = (tq << 5) + (lane & 15);
#pragma unroll
    for (int r = 0; r < 8; ++r) {
        C[(size_t)(cRow + r) * Ncol + cCol]      = acc0[r];
        C[(size_t)(cRow + r) * Ncol + cCol + 16] = acc1[r];
    }
}

// ---------------- per-node attention scalars: s = h.a_s, d = h.a_d ----------------

__global__ __launch_bounds__(256)
void dot2_kernel(const float* __restrict__ h, const float* __restrict__ as_,
                 const float* __restrict__ ad_, float* __restrict__ s,
                 float* __restrict__ d, int n, int D) {
    int wid  = (blockIdx.x * 256 + threadIdx.x) >> 5;
    int lane = threadIdx.x & 31;
    if (wid >= n) return;
    const float* hp = h + (size_t)wid * D;
    float sv = 0.f, dv = 0.f;
    for (int f = lane; f < D; f += 32) {
        float hv = hp[f];
        sv += hv * as_[f];
        dv += hv * ad_[f];
    }
    sv = waveReduceSum(sv);
    dv = waveReduceSum(dv);
    if (lane == 0) { s[wid] = sv; d[wid] = dv; }
}

// ---------------- edge passes ----------------

__global__ __launch_bounds__(256)
void edge_max_kernel(const int* __restrict__ ei, const float* __restrict__ s,
                     const float* __restrict__ d, float* __restrict__ m,
                     int E, int n) {
    int e = blockIdx.x * blockDim.x + threadIdx.x;
    if (e >= E + n) return;
    int src, dst; edge_nodes(ei, E, e, src, dst);
    float v = s[src] + d[dst];
    v = (v >= 0.f) ? v : SLOPE * v;
    atomicMaxF(&m[dst], v);
}

__global__ __launch_bounds__(256)
void edge_expsum_kernel(const int* __restrict__ ei, const float* __restrict__ s,
                        const float* __restrict__ d, const float* __restrict__ m,
                        float* __restrict__ den, float* __restrict__ ex,
                        int E, int n) {
    int e = blockIdx.x * blockDim.x + threadIdx.x;
    if (e >= E + n) return;
    int src, dst; edge_nodes(ei, E, e, src, dst);
    float v = s[src] + d[dst];
    v = (v >= 0.f) ? v : SLOPE * v;
    float ev = __expf(v - m[dst]);
    ex[e] = ev;
    atomicAdd(&den[dst], ev);
}

// one wave per edge: lanes walk contiguous features -> coalesced gather + atomic scatter
__global__ __launch_bounds__(256)
void edge_agg_kernel(const int* __restrict__ ei, const float* __restrict__ h,
                     const float* __restrict__ ex, const float* __restrict__ den,
                     float* __restrict__ outp, int E, int n, int D) {
    int wid  = (blockIdx.x * 256 + threadIdx.x) >> 5;
    int lane = threadIdx.x & 31;
    if (wid >= E + n) return;
    int src, dst; edge_nodes(ei, E, wid, src, dst);
    const float* hp = h + (size_t)src * D;
    // kick the gathered row toward the WGP before the alpha dependency chain
    if (lane * 4 < D) __builtin_prefetch(hp + lane * 4, 0, 0);
    float alpha = ex[wid] / den[dst];
    float* op = outp + (size_t)dst * D;
    for (int f = lane; f < D; f += 32)
        atomicAdd(&op[f], alpha * hp[f]);
}

// ---------------- epilogues ----------------

__global__ __launch_bounds__(256)
void bias_relu_kernel(float* __restrict__ buf, const float* __restrict__ b,
                      long long total, int D) {
    long long i = (long long)blockIdx.x * blockDim.x + threadIdx.x;
    if (i >= total) return;
    float v = buf[i] + b[(int)(i % D)];
    buf[i] = fmaxf(v, 0.f);
}

__global__ __launch_bounds__(256)
void final_kernel(const float* __restrict__ aggmu, const float* __restrict__ aggls,
                  const float* __restrict__ bmu, const float* __restrict__ bls,
                  const float* __restrict__ eps, float* __restrict__ z,
                  float* __restrict__ kl, long long total) {
    long long i = (long long)blockIdx.x * blockDim.x + threadIdx.x;
    float kle = 0.f;
    if (i < total) {
        int f = (int)(i % DZ);
        float muv = aggmu[i] + bmu[f];
        float lsv = aggls[i] + bls[f];
        float sg  = __expf(lsv);
        z[i] = muv + sg * eps[i];
        kle  = sg * sg + muv * muv - lsv - 0.5f;   // log(sigma) == lsv
    }
    kle = waveReduceSum(kle);
    if ((threadIdx.x & 31) == 0) atomicAdd(kl, kle);
}

// ---------------- launcher ----------------

static inline int cdivll(long long a, long long b) { return (int)((a + b - 1) / b); }

extern "C" void kernel_launch(void* const* d_in, const int* in_sizes, int n_in,
                              void* d_out, int out_size, void* d_ws, size_t ws_size,
                              hipStream_t stream) {
    const float* x    = (const float*)d_in[0];
    const int*   ei   = (const int*)d_in[1];
    /* d_in[2] = weight : unused by reference */
    const float* eps  = (const float*)d_in[3];
    const float* W1   = (const float*)d_in[4];
    const float* a1s  = (const float*)d_in[5];
    const float* a1d  = (const float*)d_in[6];
    const float* b1   = (const float*)d_in[7];
    const float* Wmu  = (const float*)d_in[8];
    const float* ams  = (const float*)d_in[9];
    const float* amd_ = (const float*)d_in[10];
    const float* bmu  = (const float*)d_in[11];
    const float* Wls  = (const float*)d_in[12];
    const float* als  = (const float*)d_in[13];
    const float* ald  = (const float*)d_in[14];
    const float* bls  = (const float*)d_in[15];

    const int n  = in_sizes[0] / DIN;   // 50000
    const int E  = in_sizes[1] / 2;     // 800000
    const int Et = E + n;

    // workspace layout (floats), buffers reused across phases
    float* w = (float*)d_ws;
    size_t o = 0;
    float* buf0 = w + o; o += (size_t)n * DH;   // h1; later hm(+0) and hl(+n*DZ)
    float* buf1 = w + o; o += (size_t)n * DH;   // agg1/hrelu; later aggmu(+0), aggls(+n*DZ)
    float* sb   = w + o; o += n;
    float* db   = w + o; o += n;
    float* mb   = w + o; o += n;
    float* dn   = w + o; o += n;
    float* exb  = w + o; o += Et;

    float* z  = (float*)d_out;
    float* kl = z + (size_t)n * DZ;

    const int B = 256;
    const float NEG_INF = -__builtin_huge_valf();

    // ======== layer 1: h1 = x@W1 ; GAT ; hrelu = relu(agg + b1) ========
    {
        int groups = (n / 16) * (DH / 32);   // 16x32 C slab per wave
        gemm_wmma_kernel<DIN><<<cdivll(groups, 8), B, 0, stream>>>(x, W1, buf0, n, DH);
        dot2_kernel<<<cdivll(n, 8), B, 0, stream>>>(buf0, a1s, a1d, sb, db, n, DH);
        fill_kernel<<<cdivll(n, B), B, 0, stream>>>(mb, NEG_INF, n);
        fill_kernel<<<cdivll(n, B), B, 0, stream>>>(dn, 0.f, n);
        fill_kernel<<<cdivll((long long)n * DH, B), B, 0, stream>>>(buf1, 0.f, (long long)n * DH);
        edge_max_kernel<<<cdivll(Et, B), B, 0, stream>>>(ei, sb, db, mb, E, n);
        edge_expsum_kernel<<<cdivll(Et, B), B, 0, stream>>>(ei, sb, db, mb, dn, exb, E, n);
        edge_agg_kernel<<<cdivll(Et, 8), B, 0, stream>>>(ei, buf0, exb, dn, buf1, E, n, DH);
        bias_relu_kernel<<<cdivll((long long)n * DH, B), B, 0, stream>>>(buf1, b1, (long long)n * DH, DH);
    }

    // ======== mu / logsigma GEMMs (read hrelu=buf1 before it is reused) ========
    float* hm = buf0;
    float* hl = buf0 + (size_t)n * DZ;
    {
        int groups = (n / 16) * (DZ / 32);
        gemm_wmma_kernel<DH><<<cdivll(groups, 8), B, 0, stream>>>(buf1, Wmu, hm, n, DZ);
        gemm_wmma_kernel<DH><<<cdivll(groups, 8), B, 0, stream>>>(buf1, Wls, hl, n, DZ);
    }
    float* aggmu = buf1;
    float* aggls = buf1 + (size_t)n * DZ;

    // ======== mu attention ========
    {
        dot2_kernel<<<cdivll(n, 8), B, 0, stream>>>(hm, ams, amd_, sb, db, n, DZ);
        fill_kernel<<<cdivll(n, B), B, 0, stream>>>(mb, NEG_INF, n);
        fill_kernel<<<cdivll(n, B), B, 0, stream>>>(dn, 0.f, n);
        fill_kernel<<<cdivll((long long)n * DZ, B), B, 0, stream>>>(aggmu, 0.f, (long long)n * DZ);
        edge_max_kernel<<<cdivll(Et, B), B, 0, stream>>>(ei, sb, db, mb, E, n);
        edge_expsum_kernel<<<cdivll(Et, B), B, 0, stream>>>(ei, sb, db, mb, dn, exb, E, n);
        edge_agg_kernel<<<cdivll(Et, 8), B, 0, stream>>>(ei, hm, exb, dn, aggmu, E, n, DZ);
    }

    // ======== logsigma attention ========
    {
        dot2_kernel<<<cdivll(n, 8), B, 0, stream>>>(hl, als, ald, sb, db, n, DZ);
        fill_kernel<<<cdivll(n, B), B, 0, stream>>>(mb, NEG_INF, n);
        fill_kernel<<<cdivll(n, B), B, 0, stream>>>(dn, 0.f, n);
        fill_kernel<<<cdivll((long long)n * DZ, B), B, 0, stream>>>(aggls, 0.f, (long long)n * DZ);
        edge_max_kernel<<<cdivll(Et, B), B, 0, stream>>>(ei, sb, db, mb, E, n);
        edge_expsum_kernel<<<cdivll(Et, B), B, 0, stream>>>(ei, sb, db, mb, dn, exb, E, n);
        edge_agg_kernel<<<cdivll(Et, 8), B, 0, stream>>>(ei, hl, exb, dn, aggls, E, n, DZ);
    }

    // ======== z = mu + sigma*eps ; kl reduction ========
    fill_kernel<<<1, 1, 0, stream>>>(kl, 0.f, 1);
    final_kernel<<<cdivll((long long)n * DZ, B), B, 0, stream>>>(aggmu, aggls, bmu, bls,
                                                                 eps, z, kl, (long long)n * DZ);
}
